// mGRNDropout_27822798144050
// MI455X (gfx1250) — matrix operand order repaired and checked by reading, not compiled, not run on hardware
//
#include <hip/hip_runtime.h>
#include <hip/hip_bf16.h>

typedef __attribute__((ext_vector_type(16))) __bf16 v16bf;
typedef __attribute__((ext_vector_type(8)))  __bf16 v8bf;
typedef __attribute__((ext_vector_type(8)))  float  v8f;

// ---------------- problem constants ----------------
constexpr int Tn = 512, Bn = 128, Gn = 4, FINn = 64, Hn = 256, Jn = 1024;
constexpr int Fn  = Gn * FINn;   // 256
constexpr int H3  = 3 * Hn;      // 768
constexpr int GH  = Gn * Hn;     // 1024
// ---------------- launch shape ----------------
constexpr int NWG = 64, BLK = 256;
constexpr int WAVES = NWG * (BLK / 32);   // 512 wave32 waves
// ---------------- tiling ----------------
constexpr int MT      = Bn / 16;          // 8 row tiles
constexpr int NT_MARG = H3 / 16;          // 48
constexpr int KB_X    = FINn / 32;        // 2
constexpr int KB_H    = Hn / 32;          // 8
constexpr int KB_MARG = KB_X + KB_H;      // 10
constexpr int NT_J    = Jn / 16;          // 64
constexpr int KB_F    = Fn / 32;          // 8
constexpr int KB_J    = Jn / 32;          // 32
constexpr int FRAG    = 512;              // bf16 elems per packed 32x16 B fragment

struct P {
  const float *x, *w_ih, *w_hh, *b_ih, *u_c, *b_c, *w_z, *u_z, *b_z;
  float *out;
  __bf16 *wcat, *ucp, *uzp, *wzp;          // packed bf16 B-fragments (weights)
  __bf16 *hm_b, *hj_b, *ng_b, *xb;         // bf16 activations; xb is double-buffered
  float  *px, *ph, *hm, *hj;               // fp32 pre-activations / states
  unsigned *bar;                           // grid barrier {counter, generation}
};

// ---------------- WMMA helpers ----------------
__device__ inline v8f wmma_bf16f32(v16bf a, v16bf b, v8f c) {
  // emits v_wmma_f32_16x16x32_bf16
  return __builtin_amdgcn_wmma_f32_16x16x32_bf16(false, a, false, b, (short)0, c,
                                                 false, false);
}

// A-matrix 16x32 bf16 lane order used by this code (contiguous k, same as packed B):
// lane l: row = l&15; elems 0..15 -> K = (l>>4)*16 + e
__device__ inline v16bf load_a(const __bf16* base, int ld, int lane) {
  const int row = lane & 15;
  const int kb1 = (lane >> 4) << 4;
  const __bf16* q = base + (size_t)row * ld + kb1;
  v8bf lo = *(const v8bf*)(q);
  v8bf hi = *(const v8bf*)(q + 8);
  return __builtin_shufflevector(lo, hi, 0,1,2,3,4,5,6,7,8,9,10,11,12,13,14,15);
}

// B fragments were pre-packed so each lane reads 16 contiguous bf16 (32B)
__device__ inline v16bf load_b(const __bf16* frag, int lane) {
  return *(const v16bf*)(frag + (size_t)lane * 16);
}

// C/D layout: lane l -> col = l&15 ; VGPR v -> row = (l>>4)*8 + v
__device__ inline void store_acc(float* base, int ld, int lane, v8f acc) {
  const int col = lane & 15;
  const int r0  = (lane >> 4) << 3;
#pragma unroll
  for (int v = 0; v < 8; ++v) base[(size_t)(r0 + v) * ld + col] = acc[v];
}

__device__ inline float sigmoidf_(float v) { return 1.0f / (1.0f + __expf(-v)); }

// sense/generation grid barrier (all 64 WGs resident by construction)
__device__ inline void grid_sync(unsigned* bar) {
  __threadfence();
  __syncthreads();
  if (threadIdx.x == 0) {
    unsigned gen  = __hip_atomic_load(bar + 1, __ATOMIC_ACQUIRE, __HIP_MEMORY_SCOPE_AGENT);
    unsigned prev = __hip_atomic_fetch_add(bar, 1u, __ATOMIC_ACQ_REL, __HIP_MEMORY_SCOPE_AGENT);
    if (prev == (unsigned)(gridDim.x - 1)) {
      __hip_atomic_store(bar, 0u, __ATOMIC_RELAXED, __HIP_MEMORY_SCOPE_AGENT);
      __hip_atomic_fetch_add(bar + 1, 1u, __ATOMIC_RELEASE, __HIP_MEMORY_SCOPE_AGENT);
    } else {
      while (__hip_atomic_load(bar + 1, __ATOMIC_ACQUIRE, __HIP_MEMORY_SCOPE_AGENT) == gen)
        __builtin_amdgcn_s_sleep(8);
    }
  }
  __syncthreads();
}

// ---------------- prologue: pack weights to bf16 B-fragment layout ----------------
__global__ void mgrn_pack(P p) {
  const int tid = blockIdx.x * blockDim.x + threadIdx.x;
  const int nth = gridDim.x * blockDim.x;

  for (int i = tid; i < Gn * Bn * Hn; i += nth) { p.hm[i] = 0.f; p.hm_b[i] = (__bf16)0.0f; }
  for (int i = tid; i < Bn * Jn;      i += nth) { p.hj[i] = 0.f; p.hj_b[i] = (__bf16)0.0f; }
  for (int i = tid; i < Bn * Fn;      i += nth) p.xb[i] = (__bf16)p.x[i];   // x[t=0] -> buffer 0
  if (tid == 0) { p.bar[0] = 0u; p.bar[1] = 0u; }

  // B fragment element mapping: lane l holds col = l&15, K = (l>>4)*16 + e
  // concatenated marginal weights [w_ih | w_hh] : K_total = 64 + 256 = 320
  const int P1 = Gn * NT_MARG * KB_MARG * FRAG;
  for (int i = tid; i < P1; i += nth) {
    int e = i & 15, l = (i >> 4) & 31, q = i >> 9;
    int kb = q % KB_MARG; int q2 = q / KB_MARG;
    int nt = q2 % NT_MARG; int g = q2 / NT_MARG;
    int o = nt * 16 + (l & 15);
    int K = kb * 32 + ((l >> 4) << 4) + e;
    float v = (K < FINn) ? p.w_ih[((size_t)g * H3 + o) * FINn + K]
                         : p.w_hh[((size_t)g * H3 + o) * Hn + (K - FINn)];
    p.wcat[i] = (__bf16)v;
  }
  const int P2 = NT_J * KB_J * FRAG;            // u_c and u_z, both [1024,1024]
  for (int i = tid; i < P2; i += nth) {
    int e = i & 15, l = (i >> 4) & 31, q = i >> 9;
    int kb = q % KB_J; int nt = q / KB_J;
    int o = nt * 16 + (l & 15);
    int K = kb * 32 + ((l >> 4) << 4) + e;
    p.ucp[i] = (__bf16)p.u_c[(size_t)o * GH + K];
    p.uzp[i] = (__bf16)p.u_z[(size_t)o * Jn + K];
  }
  const int P4 = NT_J * KB_F * FRAG;            // w_z [1024,256]
  for (int i = tid; i < P4; i += nth) {
    int e = i & 15, l = (i >> 4) & 31, q = i >> 9;
    int kb = q % KB_F; int nt = q / KB_F;
    int o = nt * 16 + (l & 15);
    int K = kb * 32 + ((l >> 4) << 4) + e;
    p.wzp[i] = (__bf16)p.w_z[(size_t)o * Fn + K];
  }
}

// ---------------- persistent recurrence kernel ----------------
__global__ void __launch_bounds__(BLK, 1) mgrn_persistent(P p) {
  const int lane = threadIdx.x & 31;
  const int wave = (blockIdx.x * blockDim.x + threadIdx.x) >> 5;
  const int tid  = blockIdx.x * blockDim.x + threadIdx.x;
  const int nth  = gridDim.x * blockDim.x;

  for (int t = 0; t < Tn; ++t) {
    const __bf16* xcur = p.xb + (size_t)(t & 1) * (Bn * Fn);

    // ---- phase A: marginal pre-activations, x-part & h-part (WMMA bf16) ----
    for (int tile = wave; tile < Gn * MT * NT_MARG; tile += WAVES) {
      int g = tile / (MT * NT_MARG);
      int rem = tile % (MT * NT_MARG);
      int mt = rem / NT_MARG, nt = rem % NT_MARG;
      const __bf16* xrow = xcur + (size_t)(mt * 16) * Fn + g * FINn;
      const __bf16* hrow = p.hm_b + ((size_t)g * Bn + mt * 16) * Hn;
      const __bf16* wb   = p.wcat + (size_t)(g * NT_MARG + nt) * KB_MARG * FRAG;
      v8f ax = {};
      v8f ah = {};
#pragma unroll
      for (int kb = 0; kb < KB_X; ++kb)
        ax = wmma_bf16f32(load_a(xrow + kb * 32, Fn, lane),
                          load_b(wb + kb * FRAG, lane), ax);
#pragma unroll
      for (int kb = 0; kb < KB_H; ++kb) {
        __builtin_prefetch(wb + (KB_X + kb + 1) * FRAG, 0, 1);
        ah = wmma_bf16f32(load_a(hrow + kb * 32, Hn, lane),
                          load_b(wb + (KB_X + kb) * FRAG, lane), ah);
      }
      float* pxb = p.px + ((size_t)g * Bn + mt * 16) * H3 + nt * 16;
      float* phb = p.ph + ((size_t)g * Bn + mt * 16) * H3 + nt * 16;
      store_acc(pxb, H3, lane, ax);
      store_acc(phb, H3, lane, ah);
    }
    grid_sync(p.bar);

    // ---- phase B: elementwise GRU gating + convert x[t+1] ----
    for (int i = tid; i < Gn * Bn * Hn; i += nth) {
      int g = i >> 15;                 // Bn*Hn = 32768
      int rem = i & (Bn * Hn - 1);
      int b = rem >> 8, h = rem & (Hn - 1);
      const float* px = p.px + ((size_t)g * Bn + b) * H3;
      const float* ph = p.ph + ((size_t)g * Bn + b) * H3;
      const float* bi = p.b_ih + (size_t)g * H3;
      float r = sigmoidf_(px[h] + bi[h] + ph[h]);
      float z = sigmoidf_(px[Hn + h] + bi[Hn + h] + ph[Hn + h]);
      float n = tanhf(px[2 * Hn + h] + bi[2 * Hn + h] + r * ph[2 * Hn + h]);
      float hold = p.hm[i];
      float hnew = n + z * (hold - n);
      p.hm[i]   = hnew;
      p.hm_b[i] = (__bf16)hnew;
      p.ng_b[(size_t)b * GH + g * Hn + h] = (__bf16)n;
    }
    if (t + 1 < Tn) {
      const float* xn = p.x + (size_t)(t + 1) * Bn * Fn;
      __bf16* xo = p.xb + (size_t)((t + 1) & 1) * (Bn * Fn);
      for (int i = tid; i < Bn * Fn; i += nth) xo[i] = (__bf16)xn[i];
    }
    grid_sync(p.bar);

    // ---- phase C: joint cell, fully fused (c-pre and z-pre in one wave) ----
    for (int tile = wave; tile < MT * NT_J; tile += WAVES) {
      int mt = tile / NT_J, nt = tile % NT_J;
      const __bf16* ngrow = p.ng_b + (size_t)(mt * 16) * GH;
      const __bf16* hjrow = p.hj_b + (size_t)(mt * 16) * Jn;
      const __bf16* xrow  = xcur + (size_t)(mt * 16) * Fn;
      const __bf16* ub    = p.ucp + (size_t)(nt * KB_J) * FRAG;
      const __bf16* uzb   = p.uzp + (size_t)(nt * KB_J) * FRAG;
      const __bf16* wzb   = p.wzp + (size_t)(nt * KB_F) * FRAG;
      v8f ac = {};
      v8f az = {};
#pragma unroll
      for (int kb = 0; kb < KB_F; ++kb)
        az = wmma_bf16f32(load_a(xrow + kb * 32, Fn, lane),
                          load_b(wzb + kb * FRAG, lane), az);
      for (int kb = 0; kb < KB_J; ++kb) {
        __builtin_prefetch(ub + (kb + 1) * FRAG, 0, 1);
        ac = wmma_bf16f32(load_a(ngrow + kb * 32, GH, lane),
                          load_b(ub + kb * FRAG, lane), ac);
        az = wmma_bf16f32(load_a(hjrow + kb * 32, Jn, lane),
                          load_b(uzb + kb * FRAG, lane), az);
      }
      int col = nt * 16 + (lane & 15);
      float bc = p.b_c[col], bz = p.b_z[col];
      int r0 = mt * 16 + ((lane >> 4) << 3);
#pragma unroll
      for (int v = 0; v < 8; ++v) {
        float c  = tanhf(ac[v] + bc);
        float zj = sigmoidf_(az[v] + bz);
        size_t idx = (size_t)(r0 + v) * Jn + col;
        float hold = p.hj[idx];
        float hnew = c + zj * (hold - c);
        p.hj[idx]   = hnew;
        p.hj_b[idx] = (__bf16)hnew;
        if (t == Tn - 1) p.out[idx] = hnew;
      }
    }
    grid_sync(p.bar);
  }
}

extern "C" void kernel_launch(void* const* d_in, const int* in_sizes, int n_in,
                              void* d_out, int out_size, void* d_ws, size_t ws_size,
                              hipStream_t stream) {
  (void)in_sizes; (void)n_in; (void)out_size; (void)ws_size;
  P p{};
  p.x    = (const float*)d_in[0];
  p.w_ih = (const float*)d_in[1];
  p.w_hh = (const float*)d_in[2];
  p.b_ih = (const float*)d_in[3];
  p.u_c  = (const float*)d_in[4];
  p.b_c  = (const float*)d_in[5];
  p.w_z  = (const float*)d_in[6];
  p.u_z  = (const float*)d_in[7];
  p.b_z  = (const float*)d_in[8];
  p.out  = (float*)d_out;

  char* ws = (char*)d_ws;
  size_t off = 0;
  auto alloc = [&](size_t bytes) -> char* {
    off = (off + 255) & ~(size_t)255;
    char* q = ws + off;
    off += bytes;
    return q;
  };
  p.wcat = (__bf16*)alloc((size_t)Gn * NT_MARG * KB_MARG * FRAG * 2);
  p.ucp  = (__bf16*)alloc((size_t)NT_J * KB_J * FRAG * 2);
  p.uzp  = (__bf16*)alloc((size_t)NT_J * KB_J * FRAG * 2);
  p.wzp  = (__bf16*)alloc((size_t)NT_J * KB_F * FRAG * 2);
  p.px   = (float*) alloc((size_t)Gn * Bn * H3 * 4);
  p.ph   = (float*) alloc((size_t)Gn * Bn * H3 * 4);
  p.hm   = (float*) alloc((size_t)Gn * Bn * Hn * 4);
  p.hj   = (float*) alloc((size_t)Bn * Jn * 4);
  p.hm_b = (__bf16*)alloc((size_t)Gn * Bn * Hn * 2);
  p.hj_b = (__bf16*)alloc((size_t)Bn * Jn * 2);
  p.ng_b = (__bf16*)alloc((size_t)Bn * GH * 2);
  p.xb   = (__bf16*)alloc((size_t)2 * Bn * Fn * 2);
  p.bar  = (unsigned*)alloc(256);

  mgrn_pack<<<256, BLK, 0, stream>>>(p);
  mgrn_persistent<<<NWG, BLK, 0, stream>>>(p);
}
